// TopKPerceptronRouter_50268297232578
// MI455X (gfx1250) — matrix operand order repaired and checked by reading, not compiled, not run on hardware
//
#include <hip/hip_runtime.h>

#define D_MODEL 2048
#define N_EXPERTS 64
#define TOP_K 2
#define KC 256                      // K-chunk staged in LDS
#define LDS_STRIDE (KC + 8)         // 264 bf16 per expert row (16B aligned, bank-spread)
#define TOKENS_PER_BLOCK 128
#define LOGIT_STRIDE (TOKENS_PER_BLOCK + 4)  // 132 f32 per expert row (pad vs banks)
#define THREADS 256

typedef __attribute__((ext_vector_type(16))) __bf16         v16bf;
typedef __attribute__((ext_vector_type(8)))  float          v8f;
typedef __attribute__((ext_vector_type(16))) unsigned short v16us;
typedef __attribute__((ext_vector_type(8)))  unsigned int   v8u;

// f32 -> bf16 (RNE); pairs pattern-match to v_cvt_pk_bf16_f32.
__device__ __forceinline__ unsigned short f2bf(float f) {
    return __builtin_bit_cast(unsigned short, (__bf16)f);
}

__global__ __launch_bounds__(THREADS)
void router_topk_kernel(const float* __restrict__ x,
                        const float* __restrict__ Wg,
                        const float* __restrict__ bias,
                        int*   __restrict__ out_idx,
                        float* __restrict__ out_w) {
    // W-chunk staging buffer (bf16) reused as transposed logits after the K loop.
    __shared__ union {
        unsigned short w[N_EXPERTS * LDS_STRIDE];       // 64 x 264 ushort = 33792 B
        float logitsT[N_EXPERTS * LOGIT_STRIDE];        // 64 x 132 f32    = 33792 B
    } smem;
    __shared__ float bsh[N_EXPERTS];

    const int tid  = threadIdx.x;
    const int lane = tid & 31;
    const int wave = tid >> 5;
    const int half = lane >> 4;     // 0: lanes 0-15, 1: lanes 16-31
    const int l15  = lane & 15;
    const int t0   = blockIdx.x * TOKENS_PER_BLOCK + wave * 16;   // wave's M tile

    if (tid < N_EXPERTS) bsh[tid] = bias[tid];

    // A layout (16-bit, 16x32): both lane halves hold rows M=0..15;
    // elements 0..7 = K(half*8..+7), elements 8..15 = K(16+half*8..+7).
    const float* xrow = x + (size_t)(t0 + l15) * D_MODEL;

    v8f acc[4];
    const v8f vzero = {0.f, 0.f, 0.f, 0.f, 0.f, 0.f, 0.f, 0.f};
#pragma unroll
    for (int t = 0; t < 4; ++t) acc[t] = vzero;

    // ---- software pipeline: raw-A (f32) prefetch registers, separate from afrag ----
    const float* xp0 = xrow + half * 8;
    float4 ar0 = *(const float4*)(xp0);
    float4 ar1 = *(const float4*)(xp0 + 4);
    float4 ar2 = *(const float4*)(xp0 + 16);
    float4 ar3 = *(const float4*)(xp0 + 20);

    for (int k0 = 0; k0 < D_MODEL; k0 += KC) {
        // --- stage W[:, k0:k0+KC] into LDS as bf16: float4 load -> 2x pk-cvt -> b64 store ---
#pragma unroll 4
        for (int i = tid; i < (N_EXPERTS * KC) / 4; i += THREADS) {
            const int row  = i >> 6;          // (i*4) / KC
            const int col  = (i & 63) * 4;
            const float4 wv = *(const float4*)(Wg + (size_t)row * D_MODEL + k0 + col);
            uint2 p;
            p.x = (unsigned)f2bf(wv.x) | ((unsigned)f2bf(wv.y) << 16);
            p.y = (unsigned)f2bf(wv.z) | ((unsigned)f2bf(wv.w) << 16);
            *(uint2*)&smem.w[row * LDS_STRIDE + col] = p;
        }
        __syncthreads();

#pragma unroll
        for (int kk = 0; kk < KC; kk += 32) {
            const int k = k0 + kk;

            // ---- B fragments for ALL 4 expert tiles first (distinct regs ->
            //      compiler can issue partial dscnt waits, overlap with WMMA) ----
            uint4 bq[4][2];
#pragma unroll
            for (int t = 0; t < 4; ++t) {
                const unsigned short* wp =
                    &smem.w[(t * 16 + l15) * LDS_STRIDE + kk + half * 8];
                bq[t][0] = *(const uint4*)(wp);       // ds_load_b128
                bq[t][1] = *(const uint4*)(wp + 16);  // ds_load_b128
            }

            // ---- convert current raw A -> bf16 fragment ----
            v16us au;
            au[0]  = f2bf(ar0.x); au[1]  = f2bf(ar0.y); au[2]  = f2bf(ar0.z); au[3]  = f2bf(ar0.w);
            au[4]  = f2bf(ar1.x); au[5]  = f2bf(ar1.y); au[6]  = f2bf(ar1.z); au[7]  = f2bf(ar1.w);
            au[8]  = f2bf(ar2.x); au[9]  = f2bf(ar2.y); au[10] = f2bf(ar2.z); au[11] = f2bf(ar2.w);
            au[12] = f2bf(ar3.x); au[13] = f2bf(ar3.y); au[14] = f2bf(ar3.z); au[15] = f2bf(ar3.w);
            const v16bf afrag = __builtin_bit_cast(v16bf, au);

            // ---- prefetch next k-step's raw A (clamped in-bounds on final step);
            //      writes ar*, not afrag, so no WAR against in-flight WMMAs ----
            const int kn = k + 32;
            const float* xn = xrow + ((kn < D_MODEL) ? kn : 0) + half * 8;
            ar0 = *(const float4*)(xn);
            ar1 = *(const float4*)(xn + 4);
            ar2 = *(const float4*)(xn + 16);
            ar3 = *(const float4*)(xn + 20);

            // ---- 4 WMMAs (independent accumulators; no D->A/B hazards) ----
#pragma unroll
            for (int t = 0; t < 4; ++t) {
                const v8u bu = {bq[t][0].x, bq[t][0].y, bq[t][0].z, bq[t][0].w,
                                bq[t][1].x, bq[t][1].y, bq[t][1].z, bq[t][1].w};
                const v16bf bfrag = __builtin_bit_cast(v16bf, bu);
                acc[t] = __builtin_amdgcn_wmma_f32_16x16x32_bf16(
                    /*neg_a=*/false, afrag, /*neg_b=*/false, bfrag,
                    /*c_mod=*/(short)0, acc[t],
                    /*reuse_a=*/false, /*reuse_b=*/false);
            }
        }
        __syncthreads();   // LDS chunk consumed before restage / logits reuse
    }

    // --- transposed logits scatter: C layout lane = N col, VGPR r = M row r (+8 hi).
    //     Row = expert, col = token -> 8 consecutive floats per lane = 2x ds_store_b128.
#pragma unroll
    for (int t = 0; t < 4; ++t) {
        float* dst = &smem.logitsT[(t * 16 + l15) * LOGIT_STRIDE + wave * 16 + half * 8];
        const float4 lo = {acc[t][0], acc[t][1], acc[t][2], acc[t][3]};
        const float4 hi = {acc[t][4], acc[t][5], acc[t][6], acc[t][7]};
        *(float4*)(dst)     = lo;
        *(float4*)(dst + 4) = hi;
    }
    __syncthreads();

    // --- one thread per token: bias, full-64 softmax denominator, sorted top-2 ---
    if (tid < TOKENS_PER_BLOCK) {
        float v1 = -3.402823466e38f, v2 = -3.402823466e38f;
        int   i1 = 0, i2 = 0;
#pragma unroll 8
        for (int e = 0; e < N_EXPERTS; ++e) {
            const float v = smem.logitsT[e * LOGIT_STRIDE + tid] + bsh[e];
            if (v > v1)      { v2 = v1; i2 = i1; v1 = v; i1 = e; }
            else if (v > v2) { v2 = v;  i2 = e; }
        }
        float s = 0.0f;
#pragma unroll 8
        for (int e = 0; e < N_EXPERTS; ++e)
            s += __expf(smem.logitsT[e * LOGIT_STRIDE + tid] + bsh[e] - v1);
        const float inv = 1.0f / s;
        const float p1 = inv;                       // exp(v1 - v1) / s
        const float p2 = __expf(v2 - v1) * inv;

        const size_t g = (size_t)blockIdx.x * TOKENS_PER_BLOCK + tid;
        out_idx[g * TOP_K + 0] = i1;
        out_idx[g * TOP_K + 1] = i2;
        out_w [g * TOP_K + 0] = p1;
        out_w [g * TOP_K + 1] = p2;
    }
}

extern "C" void kernel_launch(void* const* d_in, const int* in_sizes, int n_in,
                              void* d_out, int out_size, void* d_ws, size_t ws_size,
                              hipStream_t stream) {
    const float* x = (const float*)d_in[0];
    const float* W = (const float*)d_in[1];
    const float* b = (const float*)d_in[2];

    const int n_tokens = in_sizes[0] / D_MODEL;     // 4 * 8192 = 32768

    // d_out = [top_idx (int32) | top_w (f32)] concatenated flat in return order.
    int*   out_idx = (int*)d_out;
    float* out_w   = (float*)d_out + (size_t)n_tokens * TOP_K;

    const int blocks = n_tokens / TOKENS_PER_BLOCK; // 256
    hipLaunchKernelGGL(router_topk_kernel, dim3(blocks), dim3(THREADS), 0, stream,
                       x, W, b, out_idx, out_w);
}
